// RealGINEop2_v8_Model_50929722196750
// MI455X (gfx1250) — matrix operand order repaired and checked by reading, not compiled, not run on hardware
//
#include <hip/hip_runtime.h>
#include <math.h>

// ---------------- problem constants ----------------
#define N_NODES 8192
#define E_EDGES 131072
#define T_TRI   262144
#define H_DIM   128
#define IN_DIMS 64
#define G_GR    32
#define L_LAY   2
#define RBF_N   32
#define ANG_N   16
#define TM_IN   80     // 2*RBF + ANG
#define MM_IN   416    // 3*H + RBF
#define CTX_IN  384    // 3*H
#define PI_F    3.14159265358979f

typedef _Float16 h16;
typedef __attribute__((ext_vector_type(16))) _Float16 v16h;
typedef __attribute__((ext_vector_type(8)))  _Float16 v8h;
typedef __attribute__((ext_vector_type(8)))  float    v8f;
typedef __attribute__((ext_vector_type(4)))  unsigned int u32x4;
typedef __attribute__((ext_vector_type(8)))  int          i32x8;
typedef __attribute__((ext_vector_type(4)))  int          i32x4;

#if defined(__HIP_DEVICE_COMPILE__) && __has_builtin(__builtin_amdgcn_tensor_load_to_lds)
#define TDM_OK 1
#else
#define TDM_OK 0
#endif

__device__ inline v8h zero8() {
  v8h z;
#pragma unroll
  for (int i = 0; i < 8; ++i) z[i] = (h16)0.f;
  return z;
}
// convert 8 contiguous f32 (32B-aligned) -> 8 f16, via two b128 loads
__device__ inline v8h cvt8(const float* p) {
  const float4 a = *(const float4*)p;
  const float4 b = *(const float4*)(p + 4);
  v8h r;
  r[0] = (h16)a.x; r[1] = (h16)a.y; r[2] = (h16)a.z; r[3] = (h16)a.w;
  r[4] = (h16)b.x; r[5] = (h16)b.y; r[6] = (h16)b.z; r[7] = (h16)b.w;
  return r;
}

#if TDM_OK
// Tensor Data Mover: DMA a 64-row x 32-elem f16 tile (row-major, leading dim
// lda) from global into LDS, inserting a 4-dword pad after every 16 dwords so
// LDS rows land with the 40-half stride the fragment loader expects.
// D# packing per CDNA5 ISA §8.3/8.4 (best-effort; compile-only environment).
__device__ inline void tdm_stage_A(const h16* src, unsigned lds_addr, int rows, int lda) {
  unsigned long long ga = (unsigned long long)src;
  u32x4 g0;
  g0[0] = 1u;                                            // count=1 (valid user D#)
  g0[1] = lds_addr;                                      // lds_addr [63:32]
  g0[2] = (unsigned)(ga & 0xFFFFFFFFull);                // global_addr [95:64]
  g0[3] = (unsigned)((ga >> 32) & 0x01FFFFFFull) | (2u << 30);  // addr hi + type=2
  i32x8 g1;
  g1[0] = (1 << 16) | (1 << 20) | (3 << 22) | (3 << 25); // data_size=2B, pad_en,
                                                         // pad_interval=16dw, pad=4dw
  g1[1] = (int)(((unsigned)lda & 0xFFFFu) << 16);        // tensor_dim0 lo16 @ [63:48]
  g1[2] = (int)((((unsigned)lda >> 16) & 0xFFFFu) |      // tensor_dim0 hi16
                (((unsigned)rows & 0xFFFFu) << 16));     // tensor_dim1 lo16
  g1[3] = (int)((((unsigned)rows >> 16) & 0xFFFFu) |     // tensor_dim1 hi16
                (32u << 16));                            // tile_dim0 = 32 elems
  g1[4] = 64;                                            // tile_dim1 = 64 rows
  g1[5] = lda;                                           // tensor_dim0_stride lo32
  g1[6] = 0;
  g1[7] = 0;
  i32x4 z4 = {0, 0, 0, 0};
#if __clang_major__ >= 23
  i32x8 z8 = {0, 0, 0, 0, 0, 0, 0, 0};
  __builtin_amdgcn_tensor_load_to_lds(g0, g1, z4, z4, z8, 0);
#else
  __builtin_amdgcn_tensor_load_to_lds(g0, g1, z4, z4, 0);
#endif
}
#endif

// ---------------- fused GEMM kernel ----------------
// gather modes (how 8-wide slices of A(row, k..k+7) are assembled)
enum { GA_F32ROW = 0, GA_H16ROW = 1, GA_TRI = 2, GA_EDGE = 3, GA_CTX = 4 };
// epilogues
enum { EP_SILU_H16 = 0, EP_F32 = 1, EP_F32_H16 = 2, EP_SCAT_TRI = 3, EP_SCAT_EDGE = 4 };

struct GemmP {
  // generic A sources
  const float* a_f32;  const h16* a_h16;  int lda;
  // triplet gather:  t_in = [rbf[kj] | rbf[ji] | angle_feat]
  const float* rbf;  const h16* ang;  const int* idx_kj;  const int* idx_ji;
  // edge gather:     msg_in = [s[src] | s[dst] | rbf | angle_agg]
  const h16* s_h;  const float* angle_agg;  const int* src;  const int* dst;
  // ctx gather:      ctx = [s | agg_s | |v|]
  const float* s_f32;  const float* agg_s;  const float* vnorm;
  // GEMM common
  const float* W;     // K x Nout, row-major f32 weights
  const float* bias;  // Nout
  int rows, K, Nout;
  // epilogue targets
  float* out_f32; int ldo;
  h16*   out_h16; int ldoh;
  float* scat_base;      // angle_agg (tri) or agg_s (edge)
  const int* scat_idx;   // triplet_ji or dst
  const float* dir;      // E x 3 unit edge vectors (edge scatter)
  float* agg_v;          // N x H x 3 (edge scatter)
};

// 8-wide gather: every concat-segment boundary (32/64/80/128/256/288/416) is a
// multiple of 8, so an aligned 8-slice never straddles two sources.
template <int GATHER>
__device__ inline v8h gather8(const GemmP& p, int row, int k) {
  if (GATHER == GA_F32ROW) return cvt8(&p.a_f32[(size_t)row * p.lda + k]);
  if (GATHER == GA_H16ROW) return *(const v8h*)&p.a_h16[(size_t)row * p.lda + k];
  if (GATHER == GA_TRI) {
    if (k < RBF_N)     return cvt8(&p.rbf[(size_t)p.idx_kj[row] * RBF_N + k]);
    if (k < 2 * RBF_N) return cvt8(&p.rbf[(size_t)p.idx_ji[row] * RBF_N + (k - RBF_N)]);
    if (k < TM_IN)     return *(const v8h*)&p.ang[(size_t)row * ANG_N + (k - 2 * RBF_N)];
    return zero8();
  }
  if (GATHER == GA_EDGE) {
    if (k < H_DIM)     return *(const v8h*)&p.s_h[(size_t)p.src[row] * H_DIM + k];
    if (k < 2 * H_DIM) return *(const v8h*)&p.s_h[(size_t)p.dst[row] * H_DIM + (k - H_DIM)];
    if (k < 2 * H_DIM + RBF_N) return cvt8(&p.rbf[(size_t)row * RBF_N + (k - 2 * H_DIM)]);
    return cvt8(&p.angle_agg[(size_t)row * H_DIM + (k - 2 * H_DIM - RBF_N)]);
  }
  if (GATHER == GA_CTX) {
    if (k < H_DIM)     return cvt8(&p.s_f32[(size_t)row * H_DIM + k]);
    if (k < 2 * H_DIM) return cvt8(&p.agg_s[(size_t)row * H_DIM + (k - H_DIM)]);
    return cvt8(&p.vnorm[(size_t)row * H_DIM + (k - 2 * H_DIM)]);
  }
  return zero8();
}

// A fragment (16x32 f16, ISA 7.12.2 wave32 layout), from LDS row-major ld=40:
//   lanes 0-15 : row = lane,   elems[0..7]=K0..7,  [8..15]=K16..23
//   lanes16-31 : row = lane-16, elems[0..7]=K8..15, [8..15]=K24..31
__device__ inline v16h load_frag_A(const h16* base, int lane) {
  const int row = lane & 15;
  const int k0  = (lane >> 4) * 8;
  const h16* p = base + row * 40 + k0;
  const v8h lo = *(const v8h*)p;          // 16B aligned
  const v8h hi = *(const v8h*)(p + 16);   // +32B
  return __builtin_shufflevector(lo, hi, 0, 1, 2, 3, 4, 5, 6, 7,
                                 8, 9, 10, 11, 12, 13, 14, 15);
}

// Block = 128 threads (4 x wave32). Block tile = 64 rows x 128 cols.
// Wave w owns rows [w*16, w*16+16) and all 8 column sub-tiles.
// EXACT: rows is a multiple of 64 (drop all row guards statically).
// TDMODE: A is dense row-major f16 -> stage via Tensor Data Mover.
template <int GATHER, int EPILOG, bool EXACT, bool TDMODE = false>
__global__ __launch_bounds__(128) void wmma_mlp_kernel(GemmP p) {
  __shared__ __align__(16) h16 Asub[64][40];        // 64 x 32 k-chunk (+pad)
  __shared__ __align__(32) h16 Bpack[8][32][16];    // B pre-swizzled to frag layout

  const int tid  = threadIdx.x;
  const int lane = tid & 31;
  const int wave = tid >> 5;
  const int row0 = blockIdx.x * 64;
  const int nb0  = blockIdx.y * 128;

  v8f acc[8] = {};

  for (int kc = 0; kc < p.K; kc += 32) {
    if (kc + 32 < p.K)
      __builtin_prefetch((const void*)&p.W[(size_t)(kc + 32) * p.Nout + nb0], 0, 1);

#if TDM_OK
    if (TDMODE) {
      // wave 0 kicks off the DMA; it overlaps with B staging below
      if (wave == 0) {
        tdm_stage_A(p.a_h16 + (size_t)row0 * p.lda + kc,
                    (unsigned)(unsigned long long)(const void*)&Asub[0][0],
                    p.rows, p.lda);
      }
    } else
#endif
    {
      // ---- stage A: 64x32 in 8-wide groups (256 groups, 2 per thread) ----
#pragma unroll
      for (int i = 0; i < 2; ++i) {
        const int g = tid + i * 128;
        const int r = g >> 2, q = g & 3;
        const int row = row0 + r, k8 = kc + q * 8;
        v8h val;
        if ((EXACT || row < p.rows) && k8 < p.K) val = gather8<GATHER>(p, row, k8);
        else                                     val = zero8();
        *(v8h*)&Asub[r][q * 8] = val;             // single ds_store_b128
      }
    }
    // ---- stage B: 32x128 weights f32->f16, swizzled into fragment layout ----
    // Bpack[coltile][lane][elem]: lane = (n&15) + 16*((k>>3)&1), elem = (k&7) + 8*(k>=16)
#pragma unroll
    for (int i = 0; i < 8; ++i) {
      const int g4 = tid + i * 128;                 // 1024 float4 groups
      const int k  = g4 >> 5;                       // 0..31
      const int n0 = (g4 & 31) * 4;                 // 0..124
      const int kk = kc + k;
      float4 w;
      if (kk < p.K) w = *(const float4*)&p.W[(size_t)kk * p.Nout + nb0 + n0];
      else          w = make_float4(0.f, 0.f, 0.f, 0.f);
      const int ct = n0 >> 4;
      const int lb = (n0 & 15) + (((k >> 3) & 1) << 4);
      const int el = (k & 7) + ((k >> 4) << 3);
      Bpack[ct][lb + 0][el] = (h16)w.x;
      Bpack[ct][lb + 1][el] = (h16)w.y;
      Bpack[ct][lb + 2][el] = (h16)w.z;
      Bpack[ct][lb + 3][el] = (h16)w.w;
    }
#if TDM_OK
    if (TDMODE && wave == 0) __builtin_amdgcn_s_wait_tensorcnt(0);
#endif
    __syncthreads();
    // ---- 8 WMMAs per wave per k-chunk ----
    const v16h a = load_frag_A(&Asub[wave * 16][0], lane);
#pragma unroll
    for (int c = 0; c < 8; ++c) {
      const v16h b = *(const v16h*)&Bpack[c][lane][0];   // 2x ds_load_b128
      acc[c] = __builtin_amdgcn_wmma_f32_16x16x32_f16(
          false, a, false, b, (short)0, acc[c], false, false);
    }
    __syncthreads();
  }

  // ---- epilogue (C/D layout: lane = col, VGPR r -> row r or r+8) ----
  const int col16 = lane & 15;
  const int rbase = (lane >> 4) * 8;
  const int mrow0 = row0 + wave * 16;
  float bv[8];
#pragma unroll
  for (int c = 0; c < 8; ++c) bv[c] = p.bias ? p.bias[nb0 + c * 16 + col16] : 0.f;
#pragma unroll
  for (int r = 0; r < 8; ++r) {
    const int row = mrow0 + rbase + r;
    if (!EXACT && row >= p.rows) continue;
    int sidx = 0;
    float dx = 0.f, dy = 0.f, dz = 0.f;
    if (EPILOG == EP_SCAT_TRI || EPILOG == EP_SCAT_EDGE) sidx = p.scat_idx[row];
    if (EPILOG == EP_SCAT_EDGE && nb0 != 0) {
      dx = p.dir[(size_t)row * 3 + 0];
      dy = p.dir[(size_t)row * 3 + 1];
      dz = p.dir[(size_t)row * 3 + 2];
    }
#pragma unroll
    for (int c = 0; c < 8; ++c) {
      const int col = nb0 + c * 16 + col16;
      float val = acc[c][r] + bv[c];
      if (EPILOG == EP_SILU_H16) {
        val = val / (1.f + __expf(-val));
        p.out_h16[(size_t)row * p.ldoh + col] = (h16)val;
      } else if (EPILOG == EP_F32) {
        p.out_f32[(size_t)row * p.ldo + col] = val;
      } else if (EPILOG == EP_F32_H16) {
        p.out_f32[(size_t)row * p.ldo + col] = val;
        p.out_h16[(size_t)row * p.ldoh + col] = (h16)val;
      } else if (EPILOG == EP_SCAT_TRI) {
        atomicAdd(&p.scat_base[(size_t)sidx * H_DIM + col], val);
      } else if (EPILOG == EP_SCAT_EDGE) {
        if (nb0 == 0) {
          atomicAdd(&p.scat_base[(size_t)sidx * H_DIM + col], val);   // agg_s
        } else {
          const int h = col - H_DIM;                                   // agg_v
          atomicAdd(&p.agg_v[((size_t)sidx * H_DIM + h) * 3 + 0], val * dx);
          atomicAdd(&p.agg_v[((size_t)sidx * H_DIM + h) * 3 + 1], val * dy);
          atomicAdd(&p.agg_v[((size_t)sidx * H_DIM + h) * 3 + 2], val * dz);
        }
      }
    }
  }
}

// ---------------- elementwise kernels ----------------
__global__ void edge_geom_kernel(const float* __restrict__ pos,
                                 const float* __restrict__ edist,
                                 const int* __restrict__ ei,
                                 float* __restrict__ vec, float* __restrict__ dir,
                                 float* __restrict__ rbf) {
  int e = blockIdx.x * blockDim.x + threadIdx.x;
  if (e >= E_EDGES) return;
  int s = ei[e], d = ei[E_EDGES + e];
  float vx = pos[d*3+0] - pos[s*3+0];
  float vy = pos[d*3+1] - pos[s*3+1];
  float vz = pos[d*3+2] - pos[s*3+2];
  float dist = edist[e];
  float inv = 1.f / (dist + 1e-9f);
  vec[e*3+0] = vx; vec[e*3+1] = vy; vec[e*3+2] = vz;
  dir[e*3+0] = vx*inv; dir[e*3+1] = vy*inv; dir[e*3+2] = vz*inv;
  const float delta = 5.0f / (RBF_N - 1);
  const float gamma = 1.f / (2.f * (delta*delta + 1e-12f));
#pragma unroll
  for (int b = 0; b < RBF_N; ++b) {
    float df = dist - b * delta;
    rbf[(size_t)e * RBF_N + b] = __expf(-gamma * df * df);
  }
}

__global__ void angle_kernel(const float* __restrict__ vec,
                             const int* __restrict__ kj, const int* __restrict__ ji,
                             h16* __restrict__ ang) {
  int t = blockIdx.x * blockDim.x + threadIdx.x;
  if (t >= T_TRI) return;
  int ej = ji[t], ek = kj[t];
  float ax = vec[ej*3+0], ay = vec[ej*3+1], az = vec[ej*3+2];
  float bx = vec[ek*3+0], by = vec[ek*3+1], bz = vec[ek*3+2];
  float na = sqrtf(ax*ax + ay*ay + az*az + 1e-12f);
  float nb = sqrtf(bx*bx + by*by + bz*bz + 1e-12f);
  float cv = (ax*bx + ay*by + az*bz) / (na * nb + 1e-9f);
  cv = fminf(fmaxf(cv, -1.f + 1e-7f), 1.f - 1e-7f);
  float angle = acosf(cv);
  const float delta = PI_F / (ANG_N - 1);
  const float gamma = 1.f / (2.f * (delta*delta + 1e-12f));
#pragma unroll
  for (int b = 0; b < ANG_N; ++b) {
    float df = angle - b * delta;
    ang[(size_t)t * ANG_N + b] = (h16)__expf(-gamma * df * df);
  }
}

__global__ void vn_init_kernel(const float* __restrict__ vn0, float* __restrict__ vn) {
  int i = blockIdx.x * blockDim.x + threadIdx.x;
  if (i < G_GR * H_DIM) vn[i] = vn0[i & (H_DIM - 1)];
}

__global__ void add_broadcast_kernel(float* __restrict__ s, h16* __restrict__ s_h,
                                     const float* __restrict__ gout,
                                     const int* __restrict__ batch) {
  int i = blockIdx.x * blockDim.x + threadIdx.x;
  if (i >= N_NODES * H_DIM) return;
  int n = i >> 7, h = i & 127;
  float val = s[i] + gout[batch[n] * H_DIM + h];
  s[i] = val; s_h[i] = (h16)val;
}

__global__ void vnorm_kernel(const float* __restrict__ v, float* __restrict__ vnorm) {
  int i = blockIdx.x * blockDim.x + threadIdx.x;
  if (i >= N_NODES * H_DIM) return;
  float a = v[(size_t)i*3], b = v[(size_t)i*3+1], c = v[(size_t)i*3+2];
  vnorm[i] = sqrtf(a*a + b*b + c*c + 1e-12f);
}

__device__ inline float block_sum128(float x, float* red, int tid) {
#pragma unroll
  for (int off = 16; off > 0; off >>= 1) x += __shfl_xor(x, off);
  if ((tid & 31) == 0) red[tid >> 5] = x;
  __syncthreads();
  float t = red[0] + red[1] + red[2] + red[3];
  __syncthreads();
  return t;
}

__global__ __launch_bounds__(128)
void node_update_kernel(float* __restrict__ s, h16* __restrict__ s_h,
                        float* __restrict__ v,
                        const float* __restrict__ us_out, const float* __restrict__ gv_out,
                        const float* __restrict__ agg_v,
                        const float* __restrict__ lng, const float* __restrict__ lnb) {
  __shared__ float red[4];
  const int n = blockIdx.x, h = threadIdx.x;
  const size_t idx = (size_t)n * H_DIM + h;
  float sv = s[idx] + us_out[idx];
  float gate = 1.f / (1.f + __expf(-gv_out[idx]));
#pragma unroll
  for (int j = 0; j < 3; ++j) v[idx*3 + j] += gate * agg_v[idx*3 + j];
  // LayerNorm over H=128 + SiLU
  float mu = block_sum128(sv, red, h) * (1.f / H_DIM);
  float dx = sv - mu;
  float var = block_sum128(dx * dx, red, h) * (1.f / H_DIM);
  float y = dx * rsqrtf(var + 1e-5f) * lng[h] + lnb[h];
  float out = y / (1.f + __expf(-y));
  s[idx] = out; s_h[idx] = (h16)out;
}

__global__ void pool_s_kernel(const float* __restrict__ s, const int* __restrict__ batch,
                              float* __restrict__ pool) {
  int i = blockIdx.x * blockDim.x + threadIdx.x;
  if (i >= N_NODES * H_DIM) return;
  atomicAdd(&pool[batch[i >> 7] * H_DIM + (i & 127)], s[i]);
}

__global__ void vn_add_kernel(float* __restrict__ vn, const float* __restrict__ t) {
  int i = blockIdx.x * blockDim.x + threadIdx.x;
  if (i < G_GR * H_DIM) vn[i] += t[i];
}

__global__ void head_pool_kernel(const float* __restrict__ s, const float* __restrict__ vnorm,
                                 const int* __restrict__ batch,
                                 float* __restrict__ add_p, float* __restrict__ counts) {
  int i = blockIdx.x * blockDim.x + threadIdx.x;
  if (i >= N_NODES * H_DIM) return;
  int n = i >> 7, h = i & 127, g = batch[n];
  atomicAdd(&add_p[g * 2 * H_DIM + h], s[i]);
  atomicAdd(&add_p[g * 2 * H_DIM + H_DIM + h], vnorm[i]);
  if (h == 0) atomicAdd(&counts[g], 1.f);
}

__global__ void build_g_kernel(const float* __restrict__ add_p, const float* __restrict__ counts,
                               const float* __restrict__ vn, float* __restrict__ gbuf) {
  int i = blockIdx.x * blockDim.x + threadIdx.x;
  if (i >= G_GR * 5 * H_DIM) return;
  int g = i / (5 * H_DIM), c = i % (5 * H_DIM);
  float val;
  if (c < 2 * H_DIM)       val = add_p[g * 2 * H_DIM + c];
  else if (c < 4 * H_DIM)  val = add_p[g * 2 * H_DIM + (c - 2 * H_DIM)] / fmaxf(counts[g], 1.f);
  else                     val = vn[g * H_DIM + (c - 4 * H_DIM)];
  gbuf[i] = val;
}

__global__ __launch_bounds__(128)
void head_final_kernel(const h16* __restrict__ hidden, const float* __restrict__ w2,
                       const float* __restrict__ b2, float* __restrict__ out) {
  __shared__ float red[4];
  const int g = blockIdx.x, h = threadIdx.x;
  float x = (float)hidden[(size_t)g * H_DIM + h] * w2[h];
  float t = block_sum128(x, red, h);
  if (h == 0) out[g] = t + b2[0];
}

// ---------------- host orchestration ----------------
extern "C" void kernel_launch(void* const* d_in, const int* in_sizes, int n_in,
                              void* d_out, int out_size, void* d_ws, size_t ws_size,
                              hipStream_t stream) {
  (void)in_sizes; (void)n_in; (void)out_size; (void)ws_size;
  const float* x         = (const float*)d_in[0];
  const float* pos       = (const float*)d_in[1];
  const float* edge_dist = (const float*)d_in[2];
  const float* node_in_w = (const float*)d_in[3];
  const float* node_in_b = (const float*)d_in[4];
  const float* tm_w1 = (const float*)d_in[5];
  const float* tm_b1 = (const float*)d_in[6];
  const float* tm_w2 = (const float*)d_in[7];
  const float* tm_b2 = (const float*)d_in[8];
  const float* mm_w1 = (const float*)d_in[9];
  const float* mm_b1 = (const float*)d_in[10];
  const float* mm_w2 = (const float*)d_in[11];
  const float* mm_b2 = (const float*)d_in[12];
  const float* us_w1 = (const float*)d_in[13];
  const float* us_b1 = (const float*)d_in[14];
  const float* us_w2 = (const float*)d_in[15];
  const float* us_b2 = (const float*)d_in[16];
  const float* gv_w1 = (const float*)d_in[17];
  const float* gv_b1 = (const float*)d_in[18];
  const float* gv_w2 = (const float*)d_in[19];
  const float* gv_b2 = (const float*)d_in[20];
  const float* ln_g  = (const float*)d_in[21];
  const float* ln_b  = (const float*)d_in[22];
  const float* vn0   = (const float*)d_in[23];
  const float* v2n_w1 = (const float*)d_in[24];
  const float* v2n_b1 = (const float*)d_in[25];
  const float* v2n_w2 = (const float*)d_in[26];
  const float* v2n_b2 = (const float*)d_in[27];
  const float* n2v_w1 = (const float*)d_in[28];
  const float* n2v_b1 = (const float*)d_in[29];
  const float* n2v_w2 = (const float*)d_in[30];
  const float* n2v_b2 = (const float*)d_in[31];
  const float* head_w1 = (const float*)d_in[32];
  const float* head_b1 = (const float*)d_in[33];
  const float* head_w2 = (const float*)d_in[34];
  const float* head_b2 = (const float*)d_in[35];
  const int* edge_index = (const int*)d_in[36];
  const int* tri_kj = (const int*)d_in[37];
  const int* tri_ji = (const int*)d_in[38];
  const int* batch  = (const int*)d_in[39];
  const int* e_src = edge_index;            // edge_index[0]
  const int* e_dst = edge_index + E_EDGES;  // edge_index[1]

  // ---- workspace carve (deterministic bump allocator) ----
  char* cur = (char*)d_ws;
  auto alloc = [&](size_t bytes) -> void* {
    void* p = (void*)cur;
    cur += (bytes + 255) & ~(size_t)255;
    return p;
  };
  float* s      = (float*)alloc((size_t)N_NODES * H_DIM * 4);
  h16*   s_h    = (h16*)  alloc((size_t)N_NODES * H_DIM * 2);
  float* v      = (float*)alloc((size_t)N_NODES * H_DIM * 3 * 4);
  float* vnorm  = (float*)alloc((size_t)N_NODES * H_DIM * 4);
  float* vecb   = (float*)alloc((size_t)E_EDGES * 3 * 4);
  float* dirb   = (float*)alloc((size_t)E_EDGES * 3 * 4);
  float* rbfb   = (float*)alloc((size_t)E_EDGES * RBF_N * 4);
  h16*   angb   = (h16*)  alloc((size_t)T_TRI * ANG_N * 2);
  h16*   hid_t  = (h16*)  alloc((size_t)T_TRI * H_DIM * 2);
  float* angagg = (float*)alloc((size_t)E_EDGES * H_DIM * 4);
  h16*   hid_e  = (h16*)  alloc((size_t)E_EDGES * H_DIM * 2);
  float* agg_s  = (float*)alloc((size_t)N_NODES * H_DIM * 4);
  float* agg_v  = (float*)alloc((size_t)N_NODES * H_DIM * 3 * 4);
  h16*   hid_u  = (h16*)  alloc((size_t)N_NODES * H_DIM * 2);
  h16*   hid_g  = (h16*)  alloc((size_t)N_NODES * H_DIM * 2);
  float* us_out = (float*)alloc((size_t)N_NODES * H_DIM * 4);
  float* gv_out = (float*)alloc((size_t)N_NODES * H_DIM * 4);
  float* vn     = (float*)alloc((size_t)G_GR * H_DIM * 4);
  h16*   hid_s  = (h16*)  alloc((size_t)G_GR * H_DIM * 2);
  float* sm_out = (float*)alloc((size_t)G_GR * H_DIM * 4);
  float* pool   = (float*)alloc((size_t)G_GR * H_DIM * 4);
  float* add_p  = (float*)alloc((size_t)G_GR * 2 * H_DIM * 4);
  float* counts = (float*)alloc((size_t)G_GR * 4);
  float* g_buf  = (float*)alloc((size_t)G_GR * 5 * H_DIM * 4);

  const dim3 blk(128);
  auto ggrid = [](int rows, int nout) {
    return dim3((unsigned)((rows + 63) / 64), (unsigned)(nout / 128));
  };
  const int NH = N_NODES * H_DIM;

  // ---- precompute ----
  (void)hipMemsetAsync(v, 0, (size_t)N_NODES * H_DIM * 3 * 4, stream);
  {
    GemmP p{}; p.a_f32 = x; p.lda = IN_DIMS;
    p.W = node_in_w; p.bias = node_in_b;
    p.rows = N_NODES; p.K = IN_DIMS; p.Nout = H_DIM;
    p.out_f32 = s; p.ldo = H_DIM; p.out_h16 = s_h; p.ldoh = H_DIM;
    wmma_mlp_kernel<GA_F32ROW, EP_F32_H16, true><<<ggrid(N_NODES, H_DIM), blk, 0, stream>>>(p);
  }
  vn_init_kernel<<<(G_GR * H_DIM + 255) / 256, 256, 0, stream>>>(vn0, vn);
  edge_geom_kernel<<<(E_EDGES + 255) / 256, 256, 0, stream>>>(pos, edge_dist, edge_index, vecb, dirb, rbfb);
  angle_kernel<<<(T_TRI + 255) / 256, 256, 0, stream>>>(vecb, tri_kj, tri_ji, angb);

  for (int l = 0; l < L_LAY; ++l) {
    const float* tw1 = tm_w1 + (size_t)l * TM_IN * H_DIM;
    const float* tb1 = tm_b1 + (size_t)l * H_DIM;
    const float* tw2 = tm_w2 + (size_t)l * H_DIM * H_DIM;
    const float* tb2 = tm_b2 + (size_t)l * H_DIM;
    const float* mw1 = mm_w1 + (size_t)l * MM_IN * H_DIM;
    const float* mb1 = mm_b1 + (size_t)l * H_DIM;
    const float* mw2 = mm_w2 + (size_t)l * H_DIM * 2 * H_DIM;
    const float* mb2 = mm_b2 + (size_t)l * 2 * H_DIM;
    const float* uw1 = us_w1 + (size_t)l * CTX_IN * H_DIM;
    const float* ub1 = us_b1 + (size_t)l * H_DIM;
    const float* uw2 = us_w2 + (size_t)l * H_DIM * H_DIM;
    const float* ub2 = us_b2 + (size_t)l * H_DIM;
    const float* gw1 = gv_w1 + (size_t)l * CTX_IN * H_DIM;
    const float* gb1 = gv_b1 + (size_t)l * H_DIM;
    const float* gw2 = gv_w2 + (size_t)l * H_DIM * H_DIM;
    const float* gb2 = gv_b2 + (size_t)l * H_DIM;
    const float* lgl = ln_g + (size_t)l * H_DIM;
    const float* lbl = ln_b + (size_t)l * H_DIM;

    // s += v2n(vn)[batch]
    {
      GemmP p{}; p.a_f32 = vn; p.lda = H_DIM;
      p.W = v2n_w1; p.bias = v2n_b1; p.rows = G_GR; p.K = H_DIM; p.Nout = H_DIM;
      p.out_h16 = hid_s; p.ldoh = H_DIM;
      wmma_mlp_kernel<GA_F32ROW, EP_SILU_H16, false><<<ggrid(G_GR, H_DIM), blk, 0, stream>>>(p);
    }
    {
      GemmP p{}; p.a_h16 = hid_s; p.lda = H_DIM;
      p.W = v2n_w2; p.bias = v2n_b2; p.rows = G_GR; p.K = H_DIM; p.Nout = H_DIM;
      p.out_f32 = sm_out; p.ldo = H_DIM;
      wmma_mlp_kernel<GA_H16ROW, EP_F32, false><<<ggrid(G_GR, H_DIM), blk, 0, stream>>>(p);
    }
    add_broadcast_kernel<<<(NH + 255) / 256, 256, 0, stream>>>(s, s_h, sm_out, batch);

    // triplet MLP (gather-fused) -> silu hidden -> scatter-fused segment_sum over ji
    {
      GemmP p{}; p.rbf = rbfb; p.ang = angb; p.idx_kj = tri_kj; p.idx_ji = tri_ji;
      p.W = tw1; p.bias = tb1; p.rows = T_TRI; p.K = TM_IN; p.Nout = H_DIM;
      p.out_h16 = hid_t; p.ldoh = H_DIM;
      wmma_mlp_kernel<GA_TRI, EP_SILU_H16, true><<<ggrid(T_TRI, H_DIM), blk, 0, stream>>>(p);
    }
    (void)hipMemsetAsync(angagg, 0, (size_t)E_EDGES * H_DIM * 4, stream);
    {
      GemmP p{}; p.a_h16 = hid_t; p.lda = H_DIM;
      p.W = tw2; p.bias = tb2; p.rows = T_TRI; p.K = H_DIM; p.Nout = H_DIM;
      p.scat_base = angagg; p.scat_idx = tri_ji;
      wmma_mlp_kernel<GA_H16ROW, EP_SCAT_TRI, true, true><<<ggrid(T_TRI, H_DIM), blk, 0, stream>>>(p);
    }

    // edge MLP (gather-fused concat K=416) -> scatter-fused agg_s / agg_v
    {
      GemmP p{}; p.s_h = s_h; p.rbf = rbfb; p.angle_agg = angagg; p.src = e_src; p.dst = e_dst;
      p.W = mw1; p.bias = mb1; p.rows = E_EDGES; p.K = MM_IN; p.Nout = H_DIM;
      p.out_h16 = hid_e; p.ldoh = H_DIM;
      wmma_mlp_kernel<GA_EDGE, EP_SILU_H16, true><<<ggrid(E_EDGES, H_DIM), blk, 0, stream>>>(p);
    }
    (void)hipMemsetAsync(agg_s, 0, (size_t)N_NODES * H_DIM * 4, stream);
    (void)hipMemsetAsync(agg_v, 0, (size_t)N_NODES * H_DIM * 3 * 4, stream);
    {
      GemmP p{}; p.a_h16 = hid_e; p.lda = H_DIM;
      p.W = mw2; p.bias = mb2; p.rows = E_EDGES; p.K = H_DIM; p.Nout = 2 * H_DIM;
      p.scat_base = agg_s; p.scat_idx = e_dst; p.dir = dirb; p.agg_v = agg_v;
      wmma_mlp_kernel<GA_H16ROW, EP_SCAT_EDGE, true, true><<<ggrid(E_EDGES, 2 * H_DIM), blk, 0, stream>>>(p);
    }

    // node update: ctx = [s | agg_s | |v|] -> us / gv MLPs -> LN+SiLU
    vnorm_kernel<<<(NH + 255) / 256, 256, 0, stream>>>(v, vnorm);
    {
      GemmP p{}; p.s_f32 = s; p.agg_s = agg_s; p.vnorm = vnorm;
      p.W = uw1; p.bias = ub1; p.rows = N_NODES; p.K = CTX_IN; p.Nout = H_DIM;
      p.out_h16 = hid_u; p.ldoh = H_DIM;
      wmma_mlp_kernel<GA_CTX, EP_SILU_H16, true><<<ggrid(N_NODES, H_DIM), blk, 0, stream>>>(p);
    }
    {
      GemmP p{}; p.a_h16 = hid_u; p.lda = H_DIM;
      p.W = uw2; p.bias = ub2; p.rows = N_NODES; p.K = H_DIM; p.Nout = H_DIM;
      p.out_f32 = us_out; p.ldo = H_DIM;
      wmma_mlp_kernel<GA_H16ROW, EP_F32, true, true><<<ggrid(N_NODES, H_DIM), blk, 0, stream>>>(p);
    }
    {
      GemmP p{}; p.s_f32 = s; p.agg_s = agg_s; p.vnorm = vnorm;
      p.W = gw1; p.bias = gb1; p.rows = N_NODES; p.K = CTX_IN; p.Nout = H_DIM;
      p.out_h16 = hid_g; p.ldoh = H_DIM;
      wmma_mlp_kernel<GA_CTX, EP_SILU_H16, true><<<ggrid(N_NODES, H_DIM), blk, 0, stream>>>(p);
    }
    {
      GemmP p{}; p.a_h16 = hid_g; p.lda = H_DIM;
      p.W = gw2; p.bias = gb2; p.rows = N_NODES; p.K = H_DIM; p.Nout = H_DIM;
      p.out_f32 = gv_out; p.ldo = H_DIM;
      wmma_mlp_kernel<GA_H16ROW, EP_F32, true, true><<<ggrid(N_NODES, H_DIM), blk, 0, stream>>>(p);
    }
    node_update_kernel<<<N_NODES, 128, 0, stream>>>(s, s_h, v, us_out, gv_out, agg_v, lgl, lbl);

    // vn += n2v(segment_sum(s, batch))
    (void)hipMemsetAsync(pool, 0, (size_t)G_GR * H_DIM * 4, stream);
    pool_s_kernel<<<(NH + 255) / 256, 256, 0, stream>>>(s, batch, pool);
    {
      GemmP p{}; p.a_f32 = pool; p.lda = H_DIM;
      p.W = n2v_w1; p.bias = n2v_b1; p.rows = G_GR; p.K = H_DIM; p.Nout = H_DIM;
      p.out_h16 = hid_s; p.ldoh = H_DIM;
      wmma_mlp_kernel<GA_F32ROW, EP_SILU_H16, false><<<ggrid(G_GR, H_DIM), blk, 0, stream>>>(p);
    }
    {
      GemmP p{}; p.a_h16 = hid_s; p.lda = H_DIM;
      p.W = n2v_w2; p.bias = n2v_b2; p.rows = G_GR; p.K = H_DIM; p.Nout = H_DIM;
      p.out_f32 = sm_out; p.ldo = H_DIM;
      wmma_mlp_kernel<GA_H16ROW, EP_F32, false><<<ggrid(G_GR, H_DIM), blk, 0, stream>>>(p);
    }
    vn_add_kernel<<<(G_GR * H_DIM + 255) / 256, 256, 0, stream>>>(vn, sm_out);
  }

  // ---- readout head ----
  vnorm_kernel<<<(NH + 255) / 256, 256, 0, stream>>>(v, vnorm);
  (void)hipMemsetAsync(add_p, 0, (size_t)G_GR * 2 * H_DIM * 4, stream);
  (void)hipMemsetAsync(counts, 0, (size_t)G_GR * 4, stream);
  head_pool_kernel<<<(NH + 255) / 256, 256, 0, stream>>>(s, vnorm, batch, add_p, counts);
  build_g_kernel<<<(G_GR * 5 * H_DIM + 255) / 256, 256, 0, stream>>>(add_p, counts, vn, g_buf);
  {
    GemmP p{}; p.a_f32 = g_buf; p.lda = 5 * H_DIM;
    p.W = head_w1; p.bias = head_b1; p.rows = G_GR; p.K = 5 * H_DIM; p.Nout = H_DIM;
    p.out_h16 = hid_s; p.ldoh = H_DIM;
    wmma_mlp_kernel<GA_F32ROW, EP_SILU_H16, false><<<ggrid(G_GR, H_DIM), blk, 0, stream>>>(p);
  }
  head_final_kernel<<<G_GR, 128, 0, stream>>>(hid_s, head_w2, head_b2, (float*)d_out);
}
